// CustomTransformerEncoderLayer_15814069584009
// MI455X (gfx1250) — matrix operand order repaired and checked by reading, not compiled, not run on hardware
//
#include <hip/hip_runtime.h>
#include <stdint.h>

// ---------------------------------------------------------------------------
// Types
// ---------------------------------------------------------------------------
typedef __attribute__((ext_vector_type(16))) __bf16 v16bf;
typedef __attribute__((ext_vector_type(8)))  __bf16 v8bf;
typedef __attribute__((ext_vector_type(8)))  float  v8f;

__device__ __forceinline__ v16bf cat16(v8bf lo, v8bf hi) {
  return __builtin_shufflevector(lo, hi, 0,1,2,3,4,5,6,7,8,9,10,11,12,13,14,15);
}

__device__ __forceinline__ v8f wmma_bf16(v16bf a, v16bf b, v8f c) {
  // D = A(16x32 bf16) * B(32x16 bf16) + C(16x16 f32)
  return __builtin_amdgcn_wmma_f32_16x16x32_bf16(false, a, false, b,
                                                 (short)0, c, false, false);
}

// ---------------------------------------------------------------------------
// Async global->LDS copy (CDNA5). The builtin takes int4-vector pointers:
//   (v4i addrspace(1)* src, v4i addrspace(3)* dst, imm offset, imm cpol)
// ---------------------------------------------------------------------------
typedef int async_v4i __attribute__((vector_size(16)));
typedef __attribute__((address_space(1))) async_v4i* async_gptr;
typedef __attribute__((address_space(3))) async_v4i* async_lptr;

__device__ __forceinline__ void cp_async16(const void* g, void* l) {
#if defined(__has_builtin) && __has_builtin(__builtin_amdgcn_global_load_async_to_lds_b128)
  __builtin_amdgcn_global_load_async_to_lds_b128(
      (async_gptr)(void*)g, (async_lptr)l, 0, 0);
#else
  *(uint4*)l = *(const uint4*)g;
#endif
}

__device__ __forceinline__ void cp_async_wait() {
#if defined(__has_builtin) && __has_builtin(__builtin_amdgcn_s_wait_asynccnt)
  __builtin_amdgcn_s_wait_asynccnt(0);
#endif
}

// ---------------------------------------------------------------------------
// fp32 -> bf16 conversion
// ---------------------------------------------------------------------------
__global__ void cvt_bf16_kernel(const float* __restrict__ in,
                                __bf16* __restrict__ out, size_t n) {
  size_t i = (size_t)blockIdx.x * blockDim.x + threadIdx.x;
  size_t stride = (size_t)gridDim.x * blockDim.x;
  for (; i < n; i += stride) out[i] = (__bf16)in[i];
}

// ---------------------------------------------------------------------------
// Tiled WMMA GEMM:  out[m,n] = sum_k A[m,k] * W[n,k]   (+ epilogue)
// Block tile 128x128, K-step 32. 256 threads = 8 waves; each wave does a
// 64x32 sub-tile = 4x2 WMMA accumulators.
// ---------------------------------------------------------------------------
enum {
  F_BIAS = 1,   // add bias[n] (f32)
  F_GELU = 2,   // exact gelu
  F_RES  = 4,   // add resid[m*N+n] (f32)
  F_OUTF = 8,   // write f32 out
  F_OUTB = 16,  // write bf16 out
  F_VT   = 32,  // write bf16 out transposed per-head: Vt[b][h][d][n]
};

#define TM 128
#define TN 128
#define TK 32

__global__ __launch_bounds__(256)
void gemm_bf16_wmma(const __bf16* __restrict__ A,   // M x K
                    const __bf16* __restrict__ W,   // N x K
                    int M, int N, int K,
                    const float* __restrict__ bias,
                    const float* __restrict__ resid,
                    float* __restrict__ outF,
                    __bf16* __restrict__ outB,
                    int flags) {
  __shared__ __bf16 As[2][TM][TK];
  __shared__ __bf16 Bs[2][TN][TK];

  const int tid  = threadIdx.x;
  const int lane = tid & 31;
  const int wid  = tid >> 5;
  const int wm   = wid >> 2;       // 0..1 (64 rows each)
  const int wn   = wid & 3;        // 0..3 (32 cols each)
  const int hl   = lane >> 4;      // lane half
  const int l16  = lane & 15;
  const int m0   = blockIdx.y * TM;
  const int n0   = blockIdx.x * TN;

  v8f acc[4][2] = {};

  // stage one (A,B) tile pair into LDS buffer `buf` at k-offset k0
  auto stage = [&](int buf, int k0) {
    for (int ph = 0; ph < 2; ++ph) {
      int idx = tid * 8 + ph * 2048;   // 4096 bf16 per tile
      int r = idx >> 5;                // row 0..127
      int c = idx & 31;                // col 0/8/16/24
      cp_async16(&A[(size_t)(m0 + r) * K + k0 + c], &As[buf][r][c]);
      cp_async16(&W[(size_t)(n0 + r) * K + k0 + c], &Bs[buf][r][c]);
    }
  };

  const int nk = K / TK;
  stage(0, 0);
  cp_async_wait();
  __syncthreads();

  for (int kt = 0; kt < nk; ++kt) {
    const int cur = kt & 1;
    if (kt + 1 < nk) stage(cur ^ 1, (kt + 1) * TK);

    // A fragments: lane = row, 16 contiguous K values (half -> +16)
    // element chunks are [d8 .. d8+7] and [d8+8 .. d8+15].
    const int d8 = hl * 16;
    v16bf afr[4];
#pragma unroll
    for (int mt = 0; mt < 4; ++mt) {
      int r = wm * 64 + mt * 16 + l16;
      v8bf lo = *(const v8bf*)&As[cur][r][d8];
      v8bf hi = *(const v8bf*)&As[cur][r][d8 + 8];
      afr[mt] = cat16(lo, hi);
    }
    // B fragments: lane = column, 16 contiguous K values (half -> +16)
    v16bf bfr[2];
#pragma unroll
    for (int nt = 0; nt < 2; ++nt) {
      int r = wn * 32 + nt * 16 + l16;
      v8bf lo = *(const v8bf*)&Bs[cur][r][hl * 16];
      v8bf hi = *(const v8bf*)&Bs[cur][r][hl * 16 + 8];
      bfr[nt] = cat16(lo, hi);
    }
#pragma unroll
    for (int mt = 0; mt < 4; ++mt)
#pragma unroll
      for (int nt = 0; nt < 2; ++nt)
        acc[mt][nt] = wmma_bf16(afr[mt], bfr[nt], acc[mt][nt]);

    cp_async_wait();
    __syncthreads();
  }

  // Epilogue. C/D layout: VGPR r holds row (r) for lanes 0-15, (r+8) for 16-31.
#pragma unroll
  for (int mt = 0; mt < 4; ++mt) {
#pragma unroll
    for (int nt = 0; nt < 2; ++nt) {
#pragma unroll
      for (int r = 0; r < 8; ++r) {
        int m = m0 + wm * 64 + mt * 16 + r + hl * 8;
        int n = n0 + wn * 32 + nt * 16 + l16;
        float v = acc[mt][nt][r];
        if (flags & F_BIAS) v += bias[n];
        if (flags & F_GELU) v = 0.5f * v * (1.0f + erff(v * 0.70710678118f));
        if (flags & F_RES)  v += resid[(size_t)m * N + n];
        if (flags & F_OUTF) outF[(size_t)m * N + n] = v;
        if (flags & F_OUTB) outB[(size_t)m * N + n] = (__bf16)v;
        if (flags & F_VT) {
          // m = b*1024 + token, n = h*64 + d  ->  Vt[((b*16+h)*64+d)*1024+tok]
          int bb = m >> 10, tok = m & 1023, h = n >> 6, d = n & 63;
          outB[((((size_t)bb * 16 + h) * 64 + d) << 10) + tok] = (__bf16)v;
        }
      }
    }
  }
}

// ---------------------------------------------------------------------------
// Flash attention with 2D-ALiBi bias.
// One wave per (b, h, 16-query-row tile); streams key tiles of 32 with
// online softmax; O accumulated via WMMA with P staged through LDS.
// Since R_LEFT == R_RIGHT == 1, bias = slope * L1dist (le-mask cancels).
// ---------------------------------------------------------------------------
__global__ __launch_bounds__(256)
void attn_wmma_kernel(const __bf16* __restrict__ Qb,   // (B*N) x 1024
                      const __bf16* __restrict__ Kb,   // (B*N) x 1024
                      const __bf16* __restrict__ Vt,   // [B][H][64][1024]
                      const int* __restrict__ coords,  // (B*N) x 2
                      __bf16* __restrict__ outB) {     // (B*N) x 1024
  __shared__ __bf16 P[8][16][32];  // per-wave P-tile staging

  const int tid  = threadIdx.x;
  const int lane = tid & 31;
  const int wid  = tid >> 5;
  const int wg   = blockIdx.x * 8 + wid;   // global wave id, 0..4095
  const int mt   = wg & 63;                // query tile (16 rows)
  const int h    = (wg >> 6) & 15;
  const int b    = wg >> 10;
  const int hl   = lane >> 4;
  const int l16  = lane & 15;

  const float slope = exp2f(-0.5f * (float)(h + 1));
  const float scale = 0.125f;              // 1/sqrt(64)
  const int   m0    = mt * 16;

  // Q fragments (16 x 64 head-dim = two K=32 A-fragments)
  v16bf qf[2];
  {
    const size_t qrow = ((size_t)(b * 1024 + m0 + l16)) * 1024 + h * 64;
#pragma unroll
    for (int c = 0; c < 2; ++c) {
      int kb = c * 32 + hl * 16;
      v8bf lo = *(const v8bf*)&Qb[qrow + kb];
      v8bf hi = *(const v8bf*)&Qb[qrow + kb + 8];
      qf[c] = cat16(lo, hi);
    }
  }

  // coords of the query rows this lane covers (row = r + 8*hl)
  float qx[8], qy[8];
#pragma unroll
  for (int r = 0; r < 8; ++r) {
    int m = m0 + r + hl * 8;
    qx[r] = (float)coords[((size_t)(b * 1024 + m)) * 2 + 0];
    qy[r] = (float)coords[((size_t)(b * 1024 + m)) * 2 + 1];
  }

  v8f o[4] = {};
  float mi[8], li[8];
#pragma unroll
  for (int r = 0; r < 8; ++r) { mi[r] = -3.0e38f; li[r] = 0.0f; }

  for (int n0 = 0; n0 < 1024; n0 += 32) {
    // ---- S = Q K^T for two 16-wide key tiles ----
    v8f s[2];
#pragma unroll
    for (int j = 0; j < 2; ++j) {
      int n = n0 + j * 16 + l16;
      size_t kr = ((size_t)(b * 1024 + n)) * 1024 + h * 64;
      v8f z = {};
      {
        int kb = hl * 16;
        v16bf bk = cat16(*(const v8bf*)&Kb[kr + kb],
                         *(const v8bf*)&Kb[kr + kb + 8]);
        s[j] = wmma_bf16(qf[0], bk, z);
      }
      {
        int kb = 32 + hl * 16;
        v16bf bk = cat16(*(const v8bf*)&Kb[kr + kb],
                         *(const v8bf*)&Kb[kr + kb + 8]);
        s[j] = wmma_bf16(qf[1], bk, s[j]);
      }
    }
    // key coords (this lane's key column per j-tile)
    float kx[2], ky[2];
#pragma unroll
    for (int j = 0; j < 2; ++j) {
      int n = n0 + j * 16 + l16;
      kx[j] = (float)coords[((size_t)(b * 1024 + n)) * 2 + 0];
      ky[j] = (float)coords[((size_t)(b * 1024 + n)) * 2 + 1];
    }

    // ---- online softmax per row ----
    float pv0[8], pv1[8];
#pragma unroll
    for (int r = 0; r < 8; ++r) {
      float s0 = s[0][r] * scale +
                 slope * (fabsf(qx[r] - kx[0]) + fabsf(qy[r] - ky[0]));
      float s1 = s[1][r] * scale +
                 slope * (fabsf(qx[r] - kx[1]) + fabsf(qy[r] - ky[1]));
      float rm = fmaxf(s0, s1);
#pragma unroll
      for (int d = 1; d < 16; d <<= 1) rm = fmaxf(rm, __shfl_xor(rm, d, 32));
      float nm   = fmaxf(mi[r], rm);
      float corr = __expf(mi[r] - nm);
      float p0 = __expf(s0 - nm);
      float p1 = __expf(s1 - nm);
      float rs = p0 + p1;
#pragma unroll
      for (int d = 1; d < 16; d <<= 1) rs += __shfl_xor(rs, d, 32);
      li[r] = li[r] * corr + rs;
      mi[r] = nm;
#pragma unroll
      for (int jd = 0; jd < 4; ++jd) o[jd][r] *= corr;
      pv0[r] = p0; pv1[r] = p1;
    }

    // ---- P (C-layout) -> LDS row-major 16x32 ----
#pragma unroll
    for (int r = 0; r < 8; ++r) {
      int row = r + hl * 8;
      P[wid][row][l16]      = (__bf16)pv0[r];
      P[wid][row][16 + l16] = (__bf16)pv1[r];
    }
    // wave-local LDS ordering + compiler barrier
    asm volatile("s_wait_dscnt 0" ::: "memory");

    // ---- P back as A-fragment (16x32) ----
    v16bf pa;
    {
      int d8 = hl * 16;
      v8bf lo = *(const v8bf*)&P[wid][l16][d8];
      v8bf hi = *(const v8bf*)&P[wid][l16][d8 + 8];
      pa = cat16(lo, hi);
    }

    // ---- O += P * V (V pre-transposed: row d contiguous over tokens) ----
#pragma unroll
    for (int jd = 0; jd < 4; ++jd) {
      int d = jd * 16 + l16;
      size_t vb = ((((size_t)b * 16 + h) * 64 + d) << 10) + n0 + hl * 16;
      v16bf vf = cat16(*(const v8bf*)&Vt[vb], *(const v8bf*)&Vt[vb + 8]);
      o[jd] = wmma_bf16(pa, vf, o[jd]);
    }
  }

  // ---- normalize + store bf16 ----
#pragma unroll
  for (int jd = 0; jd < 4; ++jd) {
#pragma unroll
    for (int r = 0; r < 8; ++r) {
      int m = m0 + r + hl * 8;
      int d = jd * 16 + l16;
      outB[((size_t)(b * 1024 + m)) * 1024 + h * 64 + d] =
          (__bf16)(o[jd][r] / li[r]);
    }
  }
}

// ---------------------------------------------------------------------------
// LayerNorm over last dim (D=1024). One block per row.
// ---------------------------------------------------------------------------
__global__ __launch_bounds__(256)
void layernorm_kernel(const float* __restrict__ x,
                      const float* __restrict__ g,
                      const float* __restrict__ be,
                      float* __restrict__ outF,
                      __bf16* __restrict__ outB, int D) {
  __shared__ float red[2][8];
  const int row = blockIdx.x;
  const float* xr = x + (size_t)row * D;
  float s = 0.f, s2 = 0.f;
  for (int c = threadIdx.x; c < D; c += 256) {
    float v = xr[c];
    s += v; s2 += v * v;
  }
  for (int d = 1; d < 32; d <<= 1) {
    s  += __shfl_xor(s, d, 32);
    s2 += __shfl_xor(s2, d, 32);
  }
  const int wid = threadIdx.x >> 5, lane = threadIdx.x & 31;
  if (lane == 0) { red[0][wid] = s; red[1][wid] = s2; }
  __syncthreads();
  if (wid == 0) {
    s = red[0][lane & 7]; s2 = red[1][lane & 7];
    for (int d = 1; d < 8; d <<= 1) {
      s  += __shfl_xor(s, d, 32);
      s2 += __shfl_xor(s2, d, 32);
    }
    if (lane == 0) { red[0][0] = s; red[1][0] = s2; }
  }
  __syncthreads();
  const float mu   = red[0][0] / (float)D;
  const float var  = red[1][0] / (float)D - mu * mu;
  const float rstd = rsqrtf(var + 1e-5f);
  for (int c = threadIdx.x; c < D; c += 256) {
    float v = (xr[c] - mu) * rstd * g[c] + be[c];
    if (outF) outF[(size_t)row * D + c] = v;
    if (outB) outB[(size_t)row * D + c] = (__bf16)v;
  }
}

// ---------------------------------------------------------------------------
// Host orchestration
// ---------------------------------------------------------------------------
extern "C" void kernel_launch(void* const* d_in, const int* in_sizes, int n_in,
                              void* d_out, int out_size, void* d_ws,
                              size_t ws_size, hipStream_t stream) {
  (void)in_sizes; (void)n_in; (void)out_size; (void)ws_size;
  const float* src    = (const float*)d_in[0];
  const int*   coords = (const int*)d_in[1];
  const float* Wq = (const float*)d_in[2];
  const float* Wk = (const float*)d_in[3];
  const float* Wv = (const float*)d_in[4];
  const float* Wo = (const float*)d_in[5];
  const float* W1 = (const float*)d_in[6];
  const float* b1 = (const float*)d_in[7];
  const float* W2 = (const float*)d_in[8];
  const float* b2 = (const float*)d_in[9];
  const float* g1  = (const float*)d_in[10];
  const float* be1 = (const float*)d_in[11];
  const float* g2  = (const float*)d_in[12];
  const float* be2 = (const float*)d_in[13];

  const int Dm = 1024, FF = 4096;
  const size_t T = (size_t)4 * 1024;        // B*N tokens
  const size_t SB = T * Dm;                 // 4M elements

  char* ws = (char*)d_ws;
  size_t off = 0;
  auto alloc = [&](size_t bytes) -> void* {
    void* p = ws + off;
    off += (bytes + 255) & ~(size_t)255;
    return p;
  };
  __bf16* srcb = (__bf16*)alloc(SB * 2);
  __bf16* Wqb  = (__bf16*)alloc((size_t)Dm * Dm * 2);
  __bf16* Wkb  = (__bf16*)alloc((size_t)Dm * Dm * 2);
  __bf16* Wvb  = (__bf16*)alloc((size_t)Dm * Dm * 2);
  __bf16* Wob  = (__bf16*)alloc((size_t)Dm * Dm * 2);
  __bf16* W1b  = (__bf16*)alloc((size_t)FF * Dm * 2);
  __bf16* W2b  = (__bf16*)alloc((size_t)Dm * FF * 2);
  __bf16* Qb   = (__bf16*)alloc(SB * 2);
  __bf16* Kb   = (__bf16*)alloc(SB * 2);
  __bf16* Vt   = (__bf16*)alloc(SB * 2);
  __bf16* attnb= (__bf16*)alloc(SB * 2);
  __bf16* x1b  = (__bf16*)alloc(SB * 2);
  __bf16* hb   = (__bf16*)alloc(T * FF * 2);
  float*  tmp1 = (float*)alloc(SB * 4);
  float*  x1   = (float*)alloc(SB * 4);
  float*  tmp2 = (float*)alloc(SB * 4);

  // 1) convert to bf16
  cvt_bf16_kernel<<<2048, 256, 0, stream>>>(src, srcb, SB);
  cvt_bf16_kernel<<<1024, 256, 0, stream>>>(Wq, Wqb, (size_t)Dm * Dm);
  cvt_bf16_kernel<<<1024, 256, 0, stream>>>(Wk, Wkb, (size_t)Dm * Dm);
  cvt_bf16_kernel<<<1024, 256, 0, stream>>>(Wv, Wvb, (size_t)Dm * Dm);
  cvt_bf16_kernel<<<1024, 256, 0, stream>>>(Wo, Wob, (size_t)Dm * Dm);
  cvt_bf16_kernel<<<2048, 256, 0, stream>>>(W1, W1b, (size_t)FF * Dm);
  cvt_bf16_kernel<<<2048, 256, 0, stream>>>(W2, W2b, (size_t)Dm * FF);

  // 2) Q/K/V projections (V written transposed per head)
  dim3 gProj(Dm / TN, T / TM);
  gemm_bf16_wmma<<<gProj, 256, 0, stream>>>(srcb, Wqb, (int)T, Dm, Dm,
                                            nullptr, nullptr, nullptr, Qb, F_OUTB);
  gemm_bf16_wmma<<<gProj, 256, 0, stream>>>(srcb, Wkb, (int)T, Dm, Dm,
                                            nullptr, nullptr, nullptr, Kb, F_OUTB);
  gemm_bf16_wmma<<<gProj, 256, 0, stream>>>(srcb, Wvb, (int)T, Dm, Dm,
                                            nullptr, nullptr, nullptr, Vt, F_VT);

  // 3) attention (4096 waves / 8 per block)
  attn_wmma_kernel<<<512, 256, 0, stream>>>(Qb, Kb, Vt, coords, attnb);

  // 4) out-projection + residual -> tmp1 (f32)
  gemm_bf16_wmma<<<gProj, 256, 0, stream>>>(attnb, Wob, (int)T, Dm, Dm,
                                            nullptr, src, tmp1, nullptr,
                                            F_RES | F_OUTF);

  // 5) LN1 -> x1 (f32) + x1b (bf16)
  layernorm_kernel<<<(int)T, 256, 0, stream>>>(tmp1, g1, be1, x1, x1b, Dm);

  // 6) FFN up + gelu -> hb (bf16)
  dim3 gUp(FF / TN, T / TM);
  gemm_bf16_wmma<<<gUp, 256, 0, stream>>>(x1b, W1b, (int)T, FF, Dm,
                                          b1, nullptr, nullptr, hb,
                                          F_BIAS | F_GELU | F_OUTB);

  // 7) FFN down + bias + residual -> tmp2 (f32)
  gemm_bf16_wmma<<<gProj, 256, 0, stream>>>(hb, W2b, (int)T, Dm, FF,
                                            b2, x1, tmp2, nullptr,
                                            F_BIAS | F_RES | F_OUTF);

  // 8) LN2 -> output (f32)
  layernorm_kernel<<<(int)T, 256, 0, stream>>>(tmp2, g2, be2,
                                               (float*)d_out, nullptr, Dm);
}
